// W4A16Qwen2UniQLMLP_83416854823036
// MI455X (gfx1250) — matrix operand order repaired and checked by reading, not compiled, not run on hardware
//
#include <hip/hip_runtime.h>
#include <hip/hip_bf16.h>
#include <math.h>
#include <stdint.h>

typedef __attribute__((ext_vector_type(16))) _Float16 v16h;
typedef __attribute__((ext_vector_type(8)))  _Float16 v8h;
typedef __attribute__((ext_vector_type(4)))  _Float16 v4h;
typedef __attribute__((ext_vector_type(2)))  _Float16 h2;
typedef __attribute__((ext_vector_type(8)))  float    v8f;
typedef __attribute__((ext_vector_type(4)))  float    f4;
typedef __attribute__((ext_vector_type(4)))  int      i4;

// Problem dims (fixed by the reference)
constexpr int GROUPSZ = 128;
constexpr int Hdim = 4096;
constexpr int Idim = 11008;
constexpr int Mdim = 8192;   // B*S

// Tiling
constexpr int BM = 64, BN = 64, BK = 64;
constexpr int LDK = BK + 8;  // 72 halves/row = 144B, 16B-aligned

// ---------------------------------------------------------------------------
// CDNA5 async global->LDS copy, saddr (GVS) form: SGPR base + 32-bit VGPR
// byte offset (ScaleFactor=1). ASYNCcnt-tracked. 16B per op.
// ---------------------------------------------------------------------------
__device__ __forceinline__ void async_copy_b128(void* lds_ptr, const void* base, uint32_t byte_off) {
  uint32_t l = (uint32_t)(uintptr_t)lds_ptr;
  asm volatile("global_load_async_to_lds_b128 %0, %1, %2"
               :: "v"(l), "v"(byte_off), "s"(base) : "memory");
}
__device__ __forceinline__ void wait_async_all() {
  asm volatile("s_wait_asynccnt 0x0" ::: "memory");
}

// ---------------------------------------------------------------------------
// WMMA fragment builders (wave32, 16x16x32 f16, CDNA5 ISA layouts)
// ---------------------------------------------------------------------------
__device__ __forceinline__ v16h frag_a(const _Float16* rowp, int kk, int lane) {
  const int klo = kk + ((lane >> 4) << 3);
  v8h lo = *(const v8h*)(rowp + klo);
  v8h hi = *(const v8h*)(rowp + klo + 16);
  return __builtin_shufflevector(lo, hi, 0,1,2,3,4,5,6,7,8,9,10,11,12,13,14,15);
}
__device__ __forceinline__ v16h frag_b(const _Float16* colp, int kk, int lane) {
  const int klo = kk + ((lane >> 4) << 3);
  v8h lo = *(const v8h*)(colp + klo);
  v8h hi = *(const v8h*)(colp + klo + 16);
  return __builtin_shufflevector(lo, hi, 0,1,2,3,4,5,6,7,8,9,10,11,12,13,14,15);
}

// Async-stage one 64x64 f16 activation tile (512 x 16B ops / 256 threads)
__device__ __forceinline__ void load_a_tile_async(_Float16 (*dst)[LDK],
                                                  const _Float16* __restrict__ src,
                                                  uint32_t m0, uint32_t k0, uint32_t K, uint32_t tid) {
#pragma unroll
  for (uint32_t it = 0; it < 2; ++it) {
    const uint32_t v = tid + it * 256u;
    const uint32_t row = v >> 3;
    const uint32_t ck  = (v & 7u) * 8u;
    const uint32_t off = ((m0 + row) * K + (k0 + ck)) * 2u;  // byte offset, < 4GB
    async_copy_b128(&dst[row][ck], src, off);
  }
}

// ---------------------------------------------------------------------------
// Weight tile pipeline: fetch (loads->regs) ... WMMA phase ... commit (LDS)
// All global addressing via uniform base + u32 offset -> saddr-form loads.
// ---------------------------------------------------------------------------
struct WFetch {
  i4 qv[4];
  float sv[4], zv[4];
};

__device__ __forceinline__ WFetch w_fetch(const int* __restrict__ q,
                                          const float* __restrict__ s,
                                          const float* __restrict__ z,
                                          uint32_t n0, uint32_t k0, uint32_t K, uint32_t tid) {
  WFetch f;
  const uint32_t ngroups = K / GROUPSZ;
#pragma unroll
  for (uint32_t it = 0; it < 4; ++it) {
    const uint32_t v   = tid + it * 256u;
    const uint32_t row = v >> 4;
    const uint32_t ck  = (v & 15u) * 4u;
    const uint32_t n = n0 + row;
    const uint32_t k = k0 + ck;
    f.qv[it] = *(const i4*)(q + (n * K + k));          // u32 element offset
    const uint32_t g = k / GROUPSZ;
    const uint32_t so = n * ngroups + g;
    f.sv[it] = s[so];
    f.zv[it] = z[so];
  }
  return f;
}

// pk_fma_f16 dequant: w = q*s + (-z*s)
__device__ __forceinline__ void w_commit(_Float16 (*dst)[LDK], const WFetch& f, uint32_t tid) {
#pragma unroll
  for (uint32_t it = 0; it < 4; ++it) {
    const uint32_t v   = tid + it * 256u;
    const uint32_t row = v >> 4;
    const uint32_t ck  = (v & 15u) * 4u;
    const _Float16 sh = (_Float16)f.sv[it];
    const _Float16 zs = (_Float16)(-f.zv[it] * f.sv[it]);
    h2 sc2 = {sh, sh}, zs2 = {zs, zs};
    h2 q01 = {(_Float16)f.qv[it].x, (_Float16)f.qv[it].y};
    h2 q23 = {(_Float16)f.qv[it].z, (_Float16)f.qv[it].w};
    h2 w01 = __builtin_elementwise_fma(q01, sc2, zs2);
    h2 w23 = __builtin_elementwise_fma(q23, sc2, zs2);
    v4h w = {w01.x, w01.y, w23.x, w23.y};
    *(v4h*)&dst[row][ck] = w;
  }
}

#define WMMA_F16(a, b, c) \
  __builtin_amdgcn_wmma_f32_16x16x32_f16(false, (a), false, (b), (short)0, (c), false, false)

// ---------------------------------------------------------------------------
// Kernel 0: x fp32 -> f16
// ---------------------------------------------------------------------------
__global__ __launch_bounds__(256) void cvt_f32_to_f16(const float* __restrict__ x,
                                                      _Float16* __restrict__ x16, int n4) {
  const uint32_t i = blockIdx.x * 256u + threadIdx.x;
  if (i < (uint32_t)n4) {
    f4 v = *(const f4*)(x + i * 4u);
    v4h h;
    h.x = (_Float16)v.x; h.y = (_Float16)v.y; h.z = (_Float16)v.z; h.w = (_Float16)v.w;
    *(v4h*)(x16 + i * 4u) = h;
  }
}

// ---------------------------------------------------------------------------
// Kernel 1: fused gate+up GEMM + SwiGLU -> h16[M, I], double-buffered pipeline
// grid = (M/BM, I/BN): M fastest -> weights stream HBM ~once, x16 L2-resident
// ---------------------------------------------------------------------------
__global__ __launch_bounds__(256) void gate_up_swiglu(
    const _Float16* __restrict__ x16,
    const int* __restrict__ qg, const int* __restrict__ qu,
    const float* __restrict__ sg, const float* __restrict__ zg,
    const float* __restrict__ su, const float* __restrict__ zu,
    _Float16* __restrict__ h16) {
  __shared__ _Float16 sA [2][BM][LDK];
  __shared__ _Float16 sBg[2][BN][LDK];
  __shared__ _Float16 sBu[2][BN][LDK];

  const uint32_t tid  = threadIdx.x;
  const uint32_t lane = tid & 31u;
  const uint32_t wave = tid >> 5;
  const uint32_t m0 = blockIdx.x * BM;
  const uint32_t n0 = blockIdx.y * BN;
  const uint32_t wn = (wave & 3u) * 16u;
  const uint32_t wm = (wave >> 2) * 32u;
  const uint32_t ar0 = wm + (lane & 15u);
  const uint32_t ar1 = wm + 16u + (lane & 15u);
  const uint32_t bc  = wn + (lane & 15u);

  v8f accg0 = {}, accg1 = {}, accu0 = {}, accu1 = {};

  // Prologue: stage tile 0 into buffer 0
  load_a_tile_async(sA[0], x16, m0, 0, Hdim, tid);
  {
    WFetch fg = w_fetch(qg, sg, zg, n0, 0, Hdim, tid);
    WFetch fu = w_fetch(qu, su, zu, n0, 0, Hdim, tid);
    w_commit(sBg[0], fg, tid);
    w_commit(sBu[0], fu, tid);
  }
  wait_async_all();
  __syncthreads();

  // Hdim/BK = 64 (even): manual 2x-unrolled ping-pong, no remainder
  for (uint32_t kt = 0; kt < Hdim; kt += 2 * BK) {
    // ---- phase 0: compute buf0, stage kt+BK into buf1 (always valid) ----
    {
      const uint32_t ktn = kt + BK;
      load_a_tile_async(sA[1], x16, m0, ktn, Hdim, tid);
      WFetch fg = w_fetch(qg, sg, zg, n0, ktn, Hdim, tid);
      WFetch fu = w_fetch(qu, su, zu, n0, ktn, Hdim, tid);
#pragma unroll
      for (int kk = 0; kk < BK; kk += 32) {
        v16h a0 = frag_a(&sA[0][ar0][0], kk, lane);
        v16h a1 = frag_a(&sA[0][ar1][0], kk, lane);
        v16h bg = frag_b(&sBg[0][bc][0], kk, lane);
        v16h bu = frag_b(&sBu[0][bc][0], kk, lane);
        accg0 = WMMA_F16(a0, bg, accg0);
        accg1 = WMMA_F16(a1, bg, accg1);
        accu0 = WMMA_F16(a0, bu, accu0);
        accu1 = WMMA_F16(a1, bu, accu1);
      }
      w_commit(sBg[1], fg, tid);
      w_commit(sBu[1], fu, tid);
      wait_async_all();
      __syncthreads();
    }
    // ---- phase 1: compute buf1, stage kt+2BK into buf0 (if valid) ----
    {
      const uint32_t ktn = kt + 2 * BK;
      const bool have = ktn < Hdim;
      WFetch fg, fu;
      if (have) {
        load_a_tile_async(sA[0], x16, m0, ktn, Hdim, tid);
        fg = w_fetch(qg, sg, zg, n0, ktn, Hdim, tid);
        fu = w_fetch(qu, su, zu, n0, ktn, Hdim, tid);
        __builtin_prefetch(qg + (n0 + (tid & 63u)) * Hdim + ktn, 0, 1);
        __builtin_prefetch(qu + (n0 + (tid & 63u)) * Hdim + ktn, 0, 1);
      }
#pragma unroll
      for (int kk = 0; kk < BK; kk += 32) {
        v16h a0 = frag_a(&sA[1][ar0][0], kk, lane);
        v16h a1 = frag_a(&sA[1][ar1][0], kk, lane);
        v16h bg = frag_b(&sBg[1][bc][0], kk, lane);
        v16h bu = frag_b(&sBu[1][bc][0], kk, lane);
        accg0 = WMMA_F16(a0, bg, accg0);
        accg1 = WMMA_F16(a1, bg, accg1);
        accu0 = WMMA_F16(a0, bu, accu0);
        accu1 = WMMA_F16(a1, bu, accu1);
      }
      if (have) {
        w_commit(sBg[0], fg, tid);
        w_commit(sBu[0], fu, tid);
      }
      wait_async_all();
      __syncthreads();
    }
  }

  // SwiGLU epilogue: h = silu(g) * u, write f16
  const uint32_t col   = n0 + wn + (lane & 15u);
  const uint32_t rbase = (lane >> 4) ? 8u : 0u;
#pragma unroll
  for (uint32_t r = 0; r < 8; ++r) {
    {
      const float g = accg0[r], u = accu0[r];
      const float hv = (g / (1.0f + __expf(-g))) * u;
      h16[(m0 + wm + rbase + r) * (uint32_t)Idim + col] = (_Float16)hv;
    }
    {
      const float g = accg1[r], u = accu1[r];
      const float hv = (g / (1.0f + __expf(-g))) * u;
      h16[(m0 + wm + 16u + rbase + r) * (uint32_t)Idim + col] = (_Float16)hv;
    }
  }
}

// ---------------------------------------------------------------------------
// Kernel 2: down GEMM out[M, H] (fp32) = h16[M, I] x dequant(qw_down)^T
// Same double-buffered pipeline; Idim/BK = 172 (even)
// ---------------------------------------------------------------------------
__global__ __launch_bounds__(256) void down_proj(
    const _Float16* __restrict__ h16,
    const int* __restrict__ qd,
    const float* __restrict__ sd, const float* __restrict__ zd,
    float* __restrict__ out) {
  __shared__ _Float16 sA[2][BM][LDK];
  __shared__ _Float16 sB[2][BN][LDK];

  const uint32_t tid  = threadIdx.x;
  const uint32_t lane = tid & 31u;
  const uint32_t wave = tid >> 5;
  const uint32_t m0 = blockIdx.x * BM;
  const uint32_t n0 = blockIdx.y * BN;
  const uint32_t wn = (wave & 3u) * 16u;
  const uint32_t wm = (wave >> 2) * 32u;
  const uint32_t ar0 = wm + (lane & 15u);
  const uint32_t ar1 = wm + 16u + (lane & 15u);
  const uint32_t bc  = wn + (lane & 15u);

  v8f acc0 = {}, acc1 = {};

  load_a_tile_async(sA[0], h16, m0, 0, Idim, tid);
  {
    WFetch fd = w_fetch(qd, sd, zd, n0, 0, Idim, tid);
    w_commit(sB[0], fd, tid);
  }
  wait_async_all();
  __syncthreads();

  for (uint32_t kt = 0; kt < Idim; kt += 2 * BK) {
    // ---- phase 0 ----
    {
      const uint32_t ktn = kt + BK;
      load_a_tile_async(sA[1], h16, m0, ktn, Idim, tid);
      WFetch fd = w_fetch(qd, sd, zd, n0, ktn, Idim, tid);
#pragma unroll
      for (int kk = 0; kk < BK; kk += 32) {
        v16h a0 = frag_a(&sA[0][ar0][0], kk, lane);
        v16h a1 = frag_a(&sA[0][ar1][0], kk, lane);
        v16h b  = frag_b(&sB[0][bc][0], kk, lane);
        acc0 = WMMA_F16(a0, b, acc0);
        acc1 = WMMA_F16(a1, b, acc1);
      }
      w_commit(sB[1], fd, tid);
      wait_async_all();
      __syncthreads();
    }
    // ---- phase 1 ----
    {
      const uint32_t ktn = kt + 2 * BK;
      const bool have = ktn < Idim;
      WFetch fd;
      if (have) {
        load_a_tile_async(sA[0], h16, m0, ktn, Idim, tid);
        fd = w_fetch(qd, sd, zd, n0, ktn, Idim, tid);
        __builtin_prefetch(qd + (n0 + (tid & 63u)) * Idim + ktn, 0, 1);
      }
#pragma unroll
      for (int kk = 0; kk < BK; kk += 32) {
        v16h a0 = frag_a(&sA[1][ar0][0], kk, lane);
        v16h a1 = frag_a(&sA[1][ar1][0], kk, lane);
        v16h b  = frag_b(&sB[1][bc][0], kk, lane);
        acc0 = WMMA_F16(a0, b, acc0);
        acc1 = WMMA_F16(a1, b, acc1);
      }
      if (have) w_commit(sB[0], fd, tid);
      wait_async_all();
      __syncthreads();
    }
  }

  const uint32_t col   = n0 + wn + (lane & 15u);
  const uint32_t rbase = (lane >> 4) ? 8u : 0u;
#pragma unroll
  for (uint32_t r = 0; r < 8; ++r) {
    out[(m0 + wm + rbase + r)       * (uint32_t)Hdim + col] = acc0[r];
    out[(m0 + wm + 16u + rbase + r) * (uint32_t)Hdim + col] = acc1[r];
  }
}

// ---------------------------------------------------------------------------
extern "C" void kernel_launch(void* const* d_in, const int* in_sizes, int n_in,
                              void* d_out, int out_size, void* d_ws, size_t ws_size,
                              hipStream_t stream) {
  const float* x  = (const float*)d_in[0];
  const int*   qg = (const int*)  d_in[1];
  const int*   qu = (const int*)  d_in[2];
  const int*   qd = (const int*)  d_in[3];
  const float* sg = (const float*)d_in[4];
  const float* zg = (const float*)d_in[5];
  const float* su = (const float*)d_in[6];
  const float* zu = (const float*)d_in[7];
  const float* sd = (const float*)d_in[8];
  const float* zd = (const float*)d_in[9];
  float* out = (float*)d_out;

  // Workspace: x16 (64MB) then h16 (172MB)
  _Float16* x16 = (_Float16*)d_ws;
  _Float16* h16 = (_Float16*)((char*)d_ws + (size_t)Mdim * Hdim * sizeof(_Float16));

  const int n4 = (Mdim * Hdim) / 4;
  cvt_f32_to_f16<<<n4 / 256, 256, 0, stream>>>(x, x16, n4);

  dim3 g1(Mdim / BM, Idim / BN);
  gate_up_swiglu<<<g1, 256, 0, stream>>>(x16, qg, qu, sg, zg, su, zu, h16);

  dim3 g2(Mdim / BM, Hdim / BN);
  down_proj<<<g2, 256, 0, stream>>>(h16, qd, sd, zd, out);
}
